// SimpleMoERouter_44547400794423
// MI455X (gfx1250) — matrix operand order repaired and checked by reading, not compiled, not run on hardware
//
#include <hip/hip_runtime.h>

// ---- CDNA5 WMMA vector types (wave32) ----
typedef _Float16 v16h __attribute__((ext_vector_type(16)));
typedef _Float16 h8   __attribute__((ext_vector_type(8)));
typedef _Float16 h2   __attribute__((ext_vector_type(2)));
typedef float    v8f  __attribute__((ext_vector_type(8)));

#define K_DIM           4096
#define N_EXPERTS       64
#define M_TILES         4              // 4 x 16 experts = 64 (M = experts)
#define N_TILES         2              // 2 x 16 tokens = 32 tokens per group
#define TOK_PER_GRP     (16 * N_TILES)
#define KSPLIT          2              // 2 waves split K -> 2048 waves total
#define KHALF           (K_DIM / KSPLIT)          // 2048
#define TGRP_PER_BLOCK  4
#define WAVES_PER_BLOCK (TGRP_PER_BLOCK * KSPLIT) // 8
#define THREADS         (32 * WAVES_PER_BLOCK)    // 256
#define NEG_INF         (-3.4e38f)

// gate_w staged in LDS (A on DScnt; LOADcnt carries only the x stream).
#define CK       256                   // K-halves per chunk (32 KB)
#define NCHUNK   (KHALF / CK)          // 8 chunk rounds per wave
#define SEGS     (CK / 8)              // 32 x 16B segments per row
#define ROW_H    (CK + 8)              // padded stride: 132 dw == 4 mod 64 banks
#define RING     4                     // B prefetch depth (K-steps ahead)
#define RED_STRIDE 66                  // reduction stride (dw) -> conflict-free

// Stage gate_w chunks (r) and (r + NCHUNK) for the two K-halves.
// 2 chunks x 2048 segs / 256 threads = 16 b128 copies per thread per round.
__device__ __forceinline__ void stage_pair(_Float16 (*dst)[N_EXPERTS * ROW_H],
                                           const _Float16* __restrict__ gw,
                                           int r, int tid)
{
#pragma unroll
    for (int h = 0; h < KSPLIT; ++h) {
        const int cg = r + h * NCHUNK;            // global chunk id
#pragma unroll
        for (int it = 0; it < (N_EXPERTS * SEGS) / THREADS; ++it) {
            int i   = tid + it * THREADS;
            int row = i >> 5;
            int seg = i & (SEGS - 1);
            h8 v = *(const h8*)(gw + (size_t)row * K_DIM + (size_t)cg * CK + seg * 8);
            *(h8*)(dst[h] + row * ROW_H + seg * 8) = v;
        }
    }
}

__global__ __launch_bounds__(THREADS)
void SimpleMoERouter_gemm_top2_kernel(const _Float16* __restrict__ x,
                                      const _Float16* __restrict__ gw,
                                      _Float16* __restrict__ wout,
                                      int* __restrict__ iout)
{
    __shared__ _Float16 smem[2][KSPLIT][N_EXPERTS * ROW_H];  // 4 x 33 KB

    const int tid  = threadIdx.x;
    const int lane = tid & 31;
    const int wave = tid >> 5;
    const int tg   = wave & (TGRP_PER_BLOCK - 1);   // token group 0..3
    const int kh   = wave >> 2;                     // K-half 0/1
    const int tBase = blockIdx.x * (TGRP_PER_BLOCK * TOK_PER_GRP) + tg * TOK_PER_GRP;

    const int hi   = lane >> 4;
    const int lo16 = lane & 15;

    // B columns for this wave's K-half (kBase folded into the pointer).
    const _Float16* bRow[N_TILES];
#pragma unroll
    for (int i = 0; i < N_TILES; ++i)
        bRow[i] = x + (size_t)(tBase + 16 * i + lo16) * K_DIM + kh * KHALF;

    v8f acc[M_TILES][N_TILES];
#pragma unroll
    for (int j = 0; j < M_TILES; ++j)
#pragma unroll
        for (int i = 0; i < N_TILES; ++i)
            acc[j][i] = (v8f)0.0f;

    // ---- B prefetch ring: fill RING steps ahead ----
    v16h ring[RING][N_TILES];
#pragma unroll
    for (int s = 0; s < RING; ++s)
#pragma unroll
        for (int i = 0; i < N_TILES; ++i)
            ring[s][i] = __builtin_nontemporal_load(
                (const v16h*)(bRow[i] + s * 32 + hi * 16));

    stage_pair(smem[0], gw, 0, tid);

    for (int c = 0; c < NCHUNK; ++c) {
        __syncthreads();                           // round c staged
        if (c + 1 < NCHUNK)
            stage_pair(smem[(c + 1) & 1], gw, c + 1, tid);

        const _Float16* lc = smem[c & 1][kh];

#pragma unroll
        for (int s = 0; s < CK / 32; ++s) {        // 8 K-steps, fully unrolled
            const int kk   = s * 32;
            const int slot = s & (RING - 1);       // identity phi across chunks

#pragma unroll
            for (int j = 0; j < M_TILES; ++j) {
                // A tile from LDS: row 16j+lo16, chunks kk+hi*16 and +8
                const _Float16* lr = lc + (16 * j + lo16) * ROW_H + kk + hi * 16;
                h8 c0 = *(const h8*)(lr);
                h8 c1 = *(const h8*)(lr + 8);
                v16h a;
#pragma unroll
                for (int e = 0; e < 8; ++e) { a[e] = c0[e]; a[8 + e] = c1[e]; }

#pragma unroll
                for (int i = 0; i < N_TILES; ++i)
                    acc[j][i] = __builtin_amdgcn_wmma_f32_16x16x32_f16(
                        false, a, false, ring[slot][i],
                        (short)0, acc[j][i], false, false);
            }

            // Refill this slot RING steps ahead (clamped inside our K-half).
            int kn = c * CK + kk + RING * 32;
            if (kn > KHALF - 32) kn = KHALF - 32;
#pragma unroll
            for (int i = 0; i < N_TILES; ++i)
                ring[slot][i] = __builtin_nontemporal_load(
                    (const v16h*)(bRow[i] + kn + hi * 16));
        }
    }

    // ---- combine the two K-halves through LDS (padded, conflict-free) ----
    __syncthreads();                               // chunk buffers now reusable
    float* red = (float*)&smem[0][0][0];
    float* p   = red + (size_t)(tg * 32 + lane) * RED_STRIDE;
    if (kh == 1) {
#pragma unroll
        for (int j = 0; j < M_TILES; ++j)
#pragma unroll
            for (int i = 0; i < N_TILES; ++i)
#pragma unroll
                for (int r = 0; r < 8; ++r)
                    p[(j * N_TILES + i) * 8 + r] = acc[j][i][r];
    }
    __syncthreads();
    if (kh != 0) return;

#pragma unroll
    for (int j = 0; j < M_TILES; ++j)
#pragma unroll
        for (int i = 0; i < N_TILES; ++i)
#pragma unroll
            for (int r = 0; r < 8; ++r)
                acc[j][i][r] += p[(j * N_TILES + i) * 8 + r];

    // C layout: lane l -> token (l&15); vgpr r of tile j -> expert 16*j + 8*hi + r.
#pragma unroll
    for (int i = 0; i < N_TILES; ++i) {
        float b1 = NEG_INF, b2 = NEG_INF;
        int   i1 = 0, i2 = 0;
#pragma unroll
        for (int j = 0; j < M_TILES; ++j)
#pragma unroll
            for (int r = 0; r < 8; ++r) {
                float v = acc[j][i][r];
                int   e = 16 * j + 8 * hi + r;
                if (v > b1)      { b2 = b1; i2 = i1; b1 = v; i1 = e; }
                else if (v > b2) { b2 = v; i2 = e; }
            }

        float o1  = __shfl_xor(b1, 16, 32);
        int   oi1 = __shfl_xor(i1, 16, 32);
        float o2  = __shfl_xor(b2, 16, 32);
        int   oi2 = __shfl_xor(i2, 16, 32);

        float t1, t2; int ti1, ti2;
        if (b1 >= o1) {
            t1 = b1; ti1 = i1;
            if (o1 > b2) { t2 = o1; ti2 = oi1; } else { t2 = b2; ti2 = i2; }
        } else {
            t1 = o1; ti1 = oi1;
            if (b1 >= o2) { t2 = b1; ti2 = i1; } else { t2 = o2; ti2 = oi2; }
        }

        float d  = __expf(t2 - t1);          // renormalized top-2 softmax
        float w2 = d / (1.0f + d);
        float w1 = 1.0f - w2;

        if (hi == 0) {
            int token = tBase + 16 * i + lo16;
            h2 w; w[0] = (_Float16)w1; w[1] = (_Float16)w2;
            *(h2*)(wout + 2 * (size_t)token) = w;
            int2 id; id.x = ti1; id.y = ti2;
            *(int2*)(iout + 2 * (size_t)token) = id;
        }
    }
}

extern "C" void kernel_launch(void* const* d_in, const int* in_sizes, int n_in,
                              void* d_out, int out_size, void* d_ws, size_t ws_size,
                              hipStream_t stream) {
    const _Float16* x  = (const _Float16*)d_in[0];   // [tokens, 4096] fp16
    const _Float16* gw = (const _Float16*)d_in[1];   // [64, 4096] fp16

    const int tokens = in_sizes[0] / K_DIM;          // 32768

    _Float16* wout = (_Float16*)d_out;
    int*      iout = (int*)(wout + (size_t)tokens * 2);

    const int blocks = tokens / (TGRP_PER_BLOCK * TOK_PER_GRP);   // 256
    SimpleMoERouter_gemm_top2_kernel<<<dim3(blocks), dim3(THREADS), 0, stream>>>(
        x, gw, wout, iout);
}
